// SABlock_63367947485676
// MI455X (gfx1250) — matrix operand order repaired and checked by reading, not compiled, not run on hardware
//
#include <hip/hip_runtime.h>

// ---------------------------------------------------------------------------
// CDNA5 (gfx1250) WMMA-based implementation of the SABlock reference.
// All heavy math goes through v_wmma_f32_16x16x32_bf16 (wave32 matrix cores).
// Block tile 128x128, 8 wave32s, each wave computes 32x64 (8 WMMA / K-step).
// Staging: uniform full-tile fast path with vectorized loads; checked path
// only on edge tiles.
// ---------------------------------------------------------------------------

typedef __attribute__((ext_vector_type(16))) __bf16 v16bf;
typedef __attribute__((ext_vector_type(8)))  float  v8f;

union FragAB { v16bf v; unsigned u[8]; };
union FragC  { v8f   v; float    f[8]; };

constexpr int BM = 128;   // block tile M (4 waves x 32)
constexpr int BN = 128;   // block tile N (2 waves x 64)
constexpr int KT = 32;    // K per WMMA issue (bf16)
constexpr int LDSK = 36;  // padded LDS row stride (multiple of 4 -> aligned 32b ops)
constexpr int NTH = 256;  // 8 wave32s

__device__ inline unsigned short f32_to_bf16(float f) {
  unsigned u = __float_as_uint(f);
  unsigned r = u + 0x7FFFu + ((u >> 16) & 1u);  // round-to-nearest-even
  return (unsigned short)(r >> 16);
}
__device__ inline unsigned pack_bf16(float a, float b) {
  return (unsigned)f32_to_bf16(a) | ((unsigned)f32_to_bf16(b) << 16);
}
__device__ inline unsigned short ld_bf16(const float* p) { return f32_to_bf16(*p); }
__device__ inline unsigned short ld_bf16(const unsigned short* p) { return *p; }
__device__ inline void st_val(unsigned short* p, float v) { *p = f32_to_bf16(v); }
__device__ inline void st_val(float* p, float v) { *p = v; }

// Load 4 consecutive elements as 2 packed bf16 pairs. Caller guarantees the
// element offset is a multiple of 4 (16B-aligned f32 / 8B-aligned bf16).
__device__ inline uint2 ld4bf(const float* p) {
  const float4 f = *reinterpret_cast<const float4*>(p);
  uint2 r;
  r.x = pack_bf16(f.x, f.y);
  r.y = pack_bf16(f.z, f.w);
  return r;
}
__device__ inline uint2 ld4bf(const unsigned short* p) {
  return *reinterpret_cast<const uint2*>(p);
}

// Stage a TILE x KT tile (logical [t][k], k contiguous in LDS) from global.
// TRANS: element at base[k*ld + t] (else base[t*ld + k]). ld must be a
// multiple of 4 and t0 a multiple of 4 for the fast path's vector loads.
template <bool TRANS, int TILE, typename T>
__device__ inline void stage_tile(const T* __restrict__ base, unsigned short* lds,
                                  int ld, int dim0, int K, int t0, int k0, int tid) {
  const bool full = (t0 + TILE <= dim0) && (k0 + KT <= K);
  if (full) {
    if (!TRANS) {
      // vectorize along k: 4 elements per iteration
      for (int idx = tid; idx < TILE * (KT / 4); idx += NTH) {
        const int mm = idx >> 3;             // KT/4 == 8 groups per row
        const int k4 = (idx & 7) << 2;
        const uint2 r = ld4bf(base + (long long)(t0 + mm) * ld + (k0 + k4));
        *reinterpret_cast<unsigned*>(lds + mm * LDSK + k4) = r.x;
        *reinterpret_cast<unsigned*>(lds + mm * LDSK + k4 + 2) = r.y;
      }
    } else {
      // vectorize along t (contiguous in memory): 4 rows per iteration
      for (int idx = tid; idx < (TILE / 4) * KT; idx += NTH) {
        const int kk = idx / (TILE / 4);
        const int m4 = (idx % (TILE / 4)) << 2;
        const uint2 r = ld4bf(base + (long long)(k0 + kk) * ld + (t0 + m4));
        lds[(m4 + 0) * LDSK + kk] = (unsigned short)r.x;
        lds[(m4 + 1) * LDSK + kk] = (unsigned short)(r.x >> 16);
        lds[(m4 + 2) * LDSK + kk] = (unsigned short)r.y;
        lds[(m4 + 3) * LDSK + kk] = (unsigned short)(r.y >> 16);
      }
    }
  } else {
    // edge tile: scalar, zero-padded
    for (int idx = tid; idx < TILE * KT; idx += NTH) {
      const int mm = idx >> 5, kk = idx & 31;
      const int mg = t0 + mm, kg = k0 + kk;
      unsigned short vb = 0;
      if (mg < dim0 && kg < K)
        vb = ld_bf16(base + (TRANS ? ((long long)kg * ld + mg) : ((long long)mg * ld + kg)));
      lds[mm * LDSK + kk] = vb;
    }
  }
}

// A fragment (16x32 bf16, MxK): lane L holds row M=L&15.
// VGPR v holds K pair { kb + (v<4 ? 2v : 16+2(v-4)), +1 }, kb = (L>=16) ? 8 : 0.
__device__ inline void load_afrag(const unsigned short* As, int mbase, int lane, FragAB& f) {
  const unsigned short* row = As + (mbase + (lane & 15)) * LDSK;
  const int hk = (lane >> 4) << 3;  // 0 or 8
#pragma unroll
  for (int v = 0; v < 8; ++v) {
    const int k = ((v < 4) ? (v << 1) : (16 + ((v - 4) << 1))) + hk;
    f.u[v] = *reinterpret_cast<const unsigned*>(row + k);
  }
}
// B fragment (32x16 bf16, KxN) staged as Bs[n][k]: lane L holds col N=L&15.
// VGPR v holds K pair { kb + (v<4 ? 2v : 16+2(v-4)), +1 }, kb = (L>=16) ? 8 : 0.
__device__ inline void load_bfrag(const unsigned short* Bs, int nbase, int lane, FragAB& f) {
  const unsigned short* row = Bs + (nbase + (lane & 15)) * LDSK;
  const int hk = (lane >> 4) << 3;  // 0 or 8
#pragma unroll
  for (int v = 0; v < 8; ++v) {
    const int k = ((v < 4) ? (v << 1) : (16 + ((v - 4) << 1))) + hk;
    f.u[v] = *reinterpret_cast<const unsigned*>(row + k);
  }
}

template <int EPI>
__device__ inline float epi_val(float x, const float* bias, int col) {
  if (EPI == 1) {
    x += bias[col];
    x = 0.5f * x * (1.0f + erff(x * 0.70710678118654752f));
  } else if (EPI == 2) {
    x += bias[col];
  }
  return x;
}

// Generic batched GEMM: C[b][m][n] = sum_k A[b][m][k] * B[b][n][k]  (B given as N x K)
// TRANSA: A element at A[k*lda + m]; TRANSB: B element at B[k*ldb + n].
// EPI: 0 = none, 1 = +bias[n] then exact GELU, 2 = +bias[n].
template <bool TRANSA, bool TRANSB, int EPI, typename TA, typename TB, typename TOUT>
__global__ __launch_bounds__(NTH) void gemm_bf16_wmma(
    const TA* __restrict__ Ag, long long sA, int lda,
    const TB* __restrict__ Bg, long long sB, int ldb,
    TOUT* __restrict__ Cg, long long sC, int ldc,
    const float* __restrict__ bias, int M, int N, int K) {
  __shared__ unsigned short As[BM * LDSK];
  __shared__ unsigned short Bs[BN * LDSK];

  const int tid = threadIdx.x;
  const int lane = tid & 31;
  const int wid = tid >> 5;
  const int wm = (wid & 3) * 32;   // wave row offset inside block tile
  const int wn = (wid >> 2) * 64;  // wave col offset inside block tile
  const int m0 = blockIdx.y * BM;
  const int n0 = blockIdx.x * BN;

  const TA* Ab = Ag + (long long)blockIdx.z * sA;
  const TB* Bb = Bg + (long long)blockIdx.z * sB;
  TOUT* Cb = Cg + (long long)blockIdx.z * sC;

  FragC acc[2][4];
#pragma unroll
  for (int i = 0; i < 2; ++i)
#pragma unroll
    for (int j = 0; j < 4; ++j)
#pragma unroll
      for (int r = 0; r < 8; ++r) acc[i][j].f[r] = 0.0f;

  for (int k0 = 0; k0 < K; k0 += KT) {
    stage_tile<TRANSA, BM>(Ab, As, lda, M, K, m0, k0, tid);
    stage_tile<TRANSB, BN>(Bb, Bs, ldb, N, K, n0, k0, tid);
    // prefetch the next K tile while this one computes
    if (tid == 0 && k0 + KT < K) {
      const long long pa = TRANSA ? ((long long)(k0 + KT) * lda + m0) : ((long long)m0 * lda + (k0 + KT));
      const long long pb = TRANSB ? ((long long)(k0 + KT) * ldb + n0) : ((long long)n0 * ldb + (k0 + KT));
      __builtin_prefetch(Ab + pa, 0, 1);
      __builtin_prefetch(Bb + pb, 0, 1);
    }
    __syncthreads();

    FragAB af[2], bf[4];
    load_afrag(As, wm + 0, lane, af[0]);
    load_afrag(As, wm + 16, lane, af[1]);
#pragma unroll
    for (int j = 0; j < 4; ++j) load_bfrag(Bs, wn + j * 16, lane, bf[j]);
#pragma unroll
    for (int i = 0; i < 2; ++i)
#pragma unroll
      for (int j = 0; j < 4; ++j)
        acc[i][j].v = __builtin_amdgcn_wmma_f32_16x16x32_bf16(
            false, af[i].v, false, bf[j].v, (short)0, acc[i][j].v, false, false);
    __syncthreads();
  }

  // ---- epilogue: C layout VGPR r -> M = r + 8*(lane>=16), N = lane&15 ----
  const int rbase = m0 + wm + ((lane >> 4) << 3);
  const int cbase = n0 + wn + (lane & 15);
  if (m0 + BM <= M && n0 + BN <= N) {
#pragma unroll
    for (int i = 0; i < 2; ++i)
#pragma unroll
      for (int j = 0; j < 4; ++j)
#pragma unroll
        for (int r = 0; r < 8; ++r) {
          const int row = rbase + i * 16 + r;
          const int col = cbase + j * 16;
          st_val(Cb + (long long)row * ldc + col, epi_val<EPI>(acc[i][j].f[r], bias, col));
        }
  } else {
#pragma unroll
    for (int i = 0; i < 2; ++i)
#pragma unroll
      for (int j = 0; j < 4; ++j)
#pragma unroll
        for (int r = 0; r < 8; ++r) {
          const int row = rbase + i * 16 + r;
          const int col = cbase + j * 16;
          if (row < M && col < N)
            st_val(Cb + (long long)row * ldc + col, epi_val<EPI>(acc[i][j].f[r], bias, col));
        }
  }
}

// y (B,512,196) f32 -> bf16 into rows [512,1024) of A0 (B,1024,196)
__global__ __launch_bounds__(NTH) void copy_y_kernel(const float* __restrict__ y,
                                                     unsigned short* __restrict__ A0) {
  const long long total = 256LL * 512 * 196;
  long long idx = (long long)blockIdx.x * NTH + threadIdx.x;
  if (idx >= total) return;
  const long long b = idx / (512 * 196);
  const long long r = idx % (512 * 196);
  A0[b * (1024 * 196) + 512 * 196 + r] = f32_to_bf16(y[idx]);
}

// Per-batch mean / rstd over the 196x196 score matrix (InstanceNorm2d(1), biased var)
__global__ __launch_bounds__(NTH) void instnorm_stats_kernel(const float* __restrict__ s,
                                                             float* __restrict__ stats) {
  const int b = blockIdx.x;
  const float* p = s + (long long)b * 38416;
  float sum = 0.f, sq = 0.f;
  for (int i = threadIdx.x; i < 38416; i += NTH) {
    const float v = p[i];
    sum += v;
    sq += v * v;
  }
  __shared__ float rs[NTH], rq[NTH];
  rs[threadIdx.x] = sum;
  rq[threadIdx.x] = sq;
  __syncthreads();
  for (int off = NTH / 2; off > 0; off >>= 1) {
    if (threadIdx.x < off) {
      rs[threadIdx.x] += rs[threadIdx.x + off];
      rq[threadIdx.x] += rq[threadIdx.x + off];
    }
    __syncthreads();
  }
  if (threadIdx.x == 0) {
    const float m = rs[0] * (1.0f / 38416.0f);
    const float var = rq[0] * (1.0f / 38416.0f) - m * m;
    stats[2 * b] = m;
    stats[2 * b + 1] = rsqrtf(fmaxf(var, 0.0f) + 1e-5f);
  }
}

// Row softmax of the instance-normalized scores -> bf16 attention weights
__global__ __launch_bounds__(NTH) void softmax_row_kernel(const float* __restrict__ s,
                                                          const float* __restrict__ stats,
                                                          unsigned short* __restrict__ a) {
  const int b = blockIdx.y, p = blockIdx.x, t = threadIdx.x;
  const float* row = s + ((long long)b * 196 + p) * 196;
  const float mean = stats[2 * b], rstd = stats[2 * b + 1];
  float v = -3.4e38f;
  if (t < 196) v = (row[t] - mean) * rstd;
  __shared__ float red[NTH];
  red[t] = v;
  __syncthreads();
  for (int off = NTH / 2; off > 0; off >>= 1) {
    if (t < off) red[t] = fmaxf(red[t], red[t + off]);
    __syncthreads();
  }
  const float mx = red[0];
  __syncthreads();
  const float e = (t < 196) ? expf(v - mx) : 0.0f;
  red[t] = e;
  __syncthreads();
  for (int off = NTH / 2; off > 0; off >>= 1) {
    if (t < off) red[t] += red[t + off];
    __syncthreads();
  }
  const float inv = 1.0f / red[0];
  if (t < 196) a[((long long)b * 196 + p) * 196 + t] = f32_to_bf16(e * inv);
}

// ---------------------------------------------------------------------------

extern "C" void kernel_launch(void* const* d_in, const int* in_sizes, int n_in,
                              void* d_out, int out_size, void* d_ws, size_t ws_size,
                              hipStream_t stream) {
  (void)in_sizes; (void)n_in; (void)out_size; (void)ws_size;

  const float* x   = (const float*)d_in[0];   // (256,192,196)
  const float* y   = (const float*)d_in[1];   // (256,512,14,14)
  const float* Wc  = (const float*)d_in[2];   // (512,192)
  const float* Wf1 = (const float*)d_in[3];   // (392,196)
  const float* bf1 = (const float*)d_in[4];
  const float* Wf2 = (const float*)d_in[5];   // (196,392)
  const float* bf2 = (const float*)d_in[6];
  const float* Wq  = (const float*)d_in[7];   // (1024,1024)
  const float* Wk  = (const float*)d_in[8];   // (196,196)
  const float* Wv  = (const float*)d_in[9];   // (1024,1024)
  const float* Wo  = (const float*)d_in[10];  // (512,1024)
  float* out = (float*)d_out;                 // (256,512,14,14) fp32

  constexpr long long NB = 256, C1 = 1024, L = 196, F = 392, D = 1024, CO = 512;
  // Workspace layout (with reuse): A0 region reused for o; H1 region reused for q|kT.
  constexpr size_t SZ_A0 = (size_t)(NB * C1 * L) * 2;  // bf16 (o has the same size)
  constexpr size_t SZ_H1 = (size_t)(NB * C1 * F) * 2;  // bf16 (holds q + kT after G3)
  constexpr size_t SZ_A1 = (size_t)(NB * C1 * L) * 2;
  constexpr size_t SZ_V  = (size_t)(NB * L * D) * 2;
  constexpr size_t SZ_S  = (size_t)(NB * L * L) * 4;   // fp32 scores
  constexpr size_t SZ_AT = (size_t)(NB * L * L) * 2;   // bf16 attention
  constexpr size_t OFF_A0 = 0;
  constexpr size_t OFF_H1 = OFF_A0 + SZ_A0;
  constexpr size_t OFF_A1 = OFF_H1 + SZ_H1;
  constexpr size_t OFF_V  = OFF_A1 + SZ_A1;
  constexpr size_t OFF_S  = OFF_V + SZ_V;
  constexpr size_t OFF_AT = OFF_S + SZ_S;
  constexpr size_t OFF_ST = OFF_AT + SZ_AT;  // + 2KB stats => ~573 MB total

  char* ws = (char*)d_ws;
  unsigned short* A0 = (unsigned short*)(ws + OFF_A0);
  unsigned short* H1 = (unsigned short*)(ws + OFF_H1);
  unsigned short* A1 = (unsigned short*)(ws + OFF_A1);
  unsigned short* q  = H1;                       // reuse H1 after G3
  unsigned short* kT = H1 + (size_t)(NB * L * D);
  unsigned short* v  = (unsigned short*)(ws + OFF_V);
  float*          s  = (float*)(ws + OFF_S);
  unsigned short* at = (unsigned short*)(ws + OFF_AT);
  float*          st = (float*)(ws + OFF_ST);
  unsigned short* o  = A0;                       // reuse A0 after G2

  const dim3 blk(NTH);
  auto grd = [](long long M, long long N) { return dim3((N + BN - 1) / BN, (M + BM - 1) / BM, 256); };

  // G1: A0[:512] = W_conv1d(512x192) @ x_b(192x196)   (B tile = x^T access)
  gemm_bf16_wmma<false, true, 0, float, float, unsigned short><<<grd(512, L), blk, 0, stream>>>(
      Wc, 0, 192, x, 192 * L, (int)L, A0, C1 * L, (int)L, nullptr, 512, (int)L, 192);
  // A0[512:] = y (bf16 copy)
  copy_y_kernel<<<(unsigned)((NB * 512 * L + NTH - 1) / NTH), blk, 0, stream>>>(y, A0);
  // G2: H1 = gelu(A0 @ Wf1^T + bf1)   (1024x196)x(196x392)
  gemm_bf16_wmma<false, false, 1, unsigned short, float, unsigned short><<<grd(C1, F), blk, 0, stream>>>(
      A0, C1 * L, (int)L, Wf1, 0, (int)L, H1, C1 * F, (int)F, bf1, (int)C1, (int)F, (int)L);
  // G3: A1 = H1 @ Wf2^T + bf2         (1024x392)x(392x196)
  gemm_bf16_wmma<false, false, 2, unsigned short, float, unsigned short><<<grd(C1, L), blk, 0, stream>>>(
      H1, C1 * F, (int)F, Wf2, 0, (int)F, A1, C1 * L, (int)L, bf2, (int)C1, (int)L, (int)F);
  // G4: q = A1^T @ Wq^T               M=196 N=1024 K=1024, A transposed access
  gemm_bf16_wmma<true, false, 0, unsigned short, float, unsigned short><<<grd(L, D), blk, 0, stream>>>(
      A1, C1 * L, (int)L, Wq, 0, (int)D, q, L * D, (int)D, nullptr, (int)L, (int)D, (int)C1);
  // G5: kT = Wk @ A1^T  -> kT[m][d] = sum_t Wk[m][t]*A1[d][t]   (k stored transposed)
  gemm_bf16_wmma<false, false, 0, float, unsigned short, unsigned short><<<grd(L, D), blk, 0, stream>>>(
      Wk, 0, (int)L, A1, C1 * L, (int)L, kT, L * D, (int)D, nullptr, (int)L, (int)D, (int)L);
  // G6: v = A1^T @ Wv^T
  gemm_bf16_wmma<true, false, 0, unsigned short, float, unsigned short><<<grd(L, D), blk, 0, stream>>>(
      A1, C1 * L, (int)L, Wv, 0, (int)D, v, L * D, (int)D, nullptr, (int)L, (int)D, (int)C1);
  // G7: s = q @ k = q @ kT^T (both K-contiguous), fp32 out
  gemm_bf16_wmma<false, false, 0, unsigned short, unsigned short, float><<<grd(L, L), blk, 0, stream>>>(
      q, L * D, (int)D, kT, L * D, (int)D, s, L * L, (int)L, nullptr, (int)L, (int)L, (int)D);
  // InstanceNorm stats + row softmax -> bf16 attention
  instnorm_stats_kernel<<<dim3(256), blk, 0, stream>>>(s, st);
  softmax_row_kernel<<<dim3((unsigned)L, 256), blk, 0, stream>>>(s, st, at);
  // G9: o = a @ v   (B tile = v^T access)
  gemm_bf16_wmma<false, true, 0, unsigned short, unsigned short, unsigned short><<<grd(L, D), blk, 0, stream>>>(
      at, L * L, (int)L, v, L * D, (int)D, o, L * D, (int)D, nullptr, (int)L, (int)D, (int)L);
  // G10: out[b][c][p] = sum_d Wo[c][d] * o[p][d]  -> directly (512,196)=(512,14,14), fp32
  gemm_bf16_wmma<false, false, 0, float, unsigned short, float><<<grd(CO, L), blk, 0, stream>>>(
      Wo, 0, (int)D, o, L * D, (int)D, out, CO * L, (int)L, nullptr, (int)CO, (int)L, (int)D);
}
